// ComboFwdVecComp_66571993088478
// MI455X (gfx1250) — compile-verified, matrix operand order checked by reading, not run on hardware
//
#include <hip/hip_runtime.h>

#define B_ 4
#define S_ 512
#define C_ 128
#define V_ 128
#define SPAD 136   // Bt row stride in bf16 elements (128 + 8 pad -> 272B, 16B aligned, bank-spread)

typedef __attribute__((ext_vector_type(16))) __bf16 v16bf;
typedef __attribute__((ext_vector_type(8)))  __bf16 v8bf;
typedef __attribute__((ext_vector_type(8)))  float  v8f;

__device__ __forceinline__ v16bf cat8(v8bf lo, v8bf hi) {
  return __builtin_shufflevector(lo, hi, 0,1,2,3,4,5,6,7,8,9,10,11,12,13,14,15);
}

__device__ __forceinline__ float fast_tanh(float x) {
  float ax = __builtin_fabsf(x);
  // e^{-2|x|} = 2^{-2*log2(e)*|x|}
  float t = __builtin_amdgcn_exp2f(-2.885390082f * ax);
  float r = (1.0f - t) * __builtin_amdgcn_rcpf(1.0f + t);
  return __builtin_copysignf(r, x);
}

// Prologue: per (b,s) row -> bf16-convert ctx row, and Q[b,s,v] = ctx_row . (W2-Wd)[v,:] + (b1+b2+bm+bd)[v]
__global__ __launch_bounds__(128) void prep_kernel(
    const float* __restrict__ ctx, const float* __restrict__ W2, const float* __restrict__ Wd,
    const float* __restrict__ b1, const float* __restrict__ b2,
    const float* __restrict__ bm, const float* __restrict__ bd,
    __bf16* __restrict__ ctxbf, float* __restrict__ Q) {
  int row = blockIdx.x;          // 0 .. B*S-1
  int t = threadIdx.x;           // 0 .. 127  (v index)
  __shared__ float sc[C_];
  const float* cr = ctx + (size_t)row * C_;
  float x = cr[t];
  sc[t] = x;
  ctxbf[(size_t)row * C_ + t] = (__bf16)x;
  __syncthreads();
  const float* w2r = W2 + (size_t)t * C_;
  const float* wdr = Wd + (size_t)t * C_;
  float q = b1[t] + b2[t] + bm[t] + bd[t];
  #pragma unroll 8
  for (int c = 0; c < C_; ++c) q += sc[c] * (w2r[c] - wdr[c]);
  Q[(size_t)row * C_ + t] = q;
}

// Main: one block per (b,i). Fused GEMM: out[b,i,j,v] = tanh( ctx[b,j,:] @ Bt + Q[b,i,v] )
// Bt[c,v] = ctx[b,i,c]*Wm[v,c] + W1[v,c] + Wd[v,c]   (bf16, stored v-major in LDS)
__global__ __launch_bounds__(256, 1) void combo_main_kernel(
    const float* __restrict__ ctx, const float* __restrict__ W1,
    const float* __restrict__ Wm, const float* __restrict__ Wd,
    const __bf16* __restrict__ ctxbf, const float* __restrict__ Q,
    float* __restrict__ out) {
  __shared__ __bf16 Bt[V_ * SPAD];
  __shared__ float ci[C_];

  const int bi = blockIdx.x;         // b*S + i
  const int b  = bi >> 9;            // / S_
  const int t  = threadIdx.x;        // 0..255

  if (t < C_) ci[t] = ctx[(size_t)bi * C_ + t];
  __syncthreads();

  // Build fused B matrix (N-major: Bt[v*SPAD + c])
  for (int idx = t; idx < V_ * C_; idx += 256) {
    int v = idx >> 7, c = idx & 127;
    float val = ci[c] * Wm[idx] + W1[idx] + Wd[idx];
    Bt[v * SPAD + c] = (__bf16)val;
  }
  __syncthreads();

  const int w    = t >> 5;           // wave 0..7
  const int lane = t & 31;
  const int half = lane >> 4;        // 0: K 0-7/16-23, 1: K 8-15/24-31
  const int ln   = lane & 15;

  const __bf16* Abase = ctxbf + (size_t)b * (S_ * C_);
  const float*  qrow  = Q + (size_t)bi * V_;
  float qv[8];
  #pragma unroll
  for (int n = 0; n < 8; ++n) qv[n] = qrow[n * 16 + ln];

  float* outb = out + (size_t)bi * (S_ * V_);   // contiguous 512x128 slab

  for (int p = 0; p < 2; ++p) {
    const int mA = w + 16 * p;       // M-tiles: p0 -> 0..15, p1 -> 16..31 (paired per wave)
    const int mB = mA + 8;

    // A fragments for both M-tiles (16-bit A layout: per lane two 8-elem runs)
    v16bf aA[4], aB[4];
    #pragma unroll
    for (int kk = 0; kk < 4; ++kk) {
      const __bf16* pa = Abase + ((size_t)(mA * 16 + ln) * C_ + kk * 32 + half * 8);
      v8bf lo = *(const v8bf*)pa;
      v8bf hi = *(const v8bf*)(pa + 16);
      aA[kk] = cat8(lo, hi);
      const __bf16* pb = Abase + ((size_t)(mB * 16 + ln) * C_ + kk * 32 + half * 8);
      lo = *(const v8bf*)pb;
      hi = *(const v8bf*)(pb + 16);
      aB[kk] = cat8(lo, hi);
    }

    v8f accA[8] = {};
    v8f accB[8] = {};

    #pragma unroll
    for (int kk = 0; kk < 4; ++kk) {
      #pragma unroll
      for (int n = 0; n < 8; ++n) {
        const __bf16* pbm = &Bt[(n * 16 + ln) * SPAD + kk * 32 + half * 8];
        v8bf lo = *(const v8bf*)pbm;
        v8bf hi = *(const v8bf*)(pbm + 16);
        v16bf bb = cat8(lo, hi);
        accA[n] = __builtin_amdgcn_wmma_f32_16x16x32_bf16(
            false, aA[kk], false, bb, (short)0, accA[n], false, false);
        accB[n] = __builtin_amdgcn_wmma_f32_16x16x32_bf16(
            false, aB[kk], false, bb, (short)0, accB[n], false, false);
      }
    }

    // Epilogue: add bias, tanh, stream out (D layout: vgpr r, lanes0-15 M=r, lanes16-31 M=r+8)
    #pragma unroll
    for (int n = 0; n < 8; ++n) {
      const int vcol = n * 16 + ln;
      #pragma unroll
      for (int r = 0; r < 8; ++r) {
        int jA = mA * 16 + r + 8 * half;
        outb[(size_t)jA * V_ + vcol] = fast_tanh(accA[n][r] + qv[n]);
        int jB = mB * 16 + r + 8 * half;
        outb[(size_t)jB * V_ + vcol] = fast_tanh(accB[n][r] + qv[n]);
      }
    }
  }
}

extern "C" void kernel_launch(void* const* d_in, const int* in_sizes, int n_in,
                              void* d_out, int out_size, void* d_ws, size_t ws_size,
                              hipStream_t stream) {
  const float* ctx = (const float*)d_in[0];
  const float* W1  = (const float*)d_in[1];
  const float* b1  = (const float*)d_in[2];
  const float* W2  = (const float*)d_in[3];
  const float* b2  = (const float*)d_in[4];
  const float* Wm  = (const float*)d_in[5];
  const float* bm  = (const float*)d_in[6];
  const float* Wd  = (const float*)d_in[7];
  const float* bd  = (const float*)d_in[8];
  float* out = (float*)d_out;

  // Workspace: [ctx in bf16: B*S*C*2 = 512KB][Q: B*S*V*4 = 1MB]
  __bf16* ctxbf = (__bf16*)d_ws;
  float*  Q     = (float*)((char*)d_ws + (size_t)B_ * S_ * C_ * sizeof(__bf16));

  prep_kernel<<<B_ * S_, 128, 0, stream>>>(ctx, W2, Wd, b1, b2, bm, bd, ctxbf, Q);
  combo_main_kernel<<<B_ * S_, 256, 0, stream>>>(ctx, W1, Wm, Wd, ctxbf, Q, out);
}
